// VectorQuantizer_17428977287171
// MI455X (gfx1250) — matrix operand order repaired and checked by reading, not compiled, not run on hardware
//
#include <hip/hip_runtime.h>
#include <hip/hip_bf16.h>
#include <math.h>

// ---------------------------------------------------------------------------
// VQ-VAE vector quantizer for gfx1250 (MI455X).
//   x: (8192, 256) f32   codebook: (8192, 256) f32
//   Distance GEMM on v_wmma_f32_16x16x32_f16; codebook tiles DMA'd into LDS
//   by the Tensor Data Mover (tensor_load_to_lds + s_wait_tensorcnt),
//   double-buffered behind the WMMA stream.
// ---------------------------------------------------------------------------

typedef __attribute__((ext_vector_type(16))) _Float16 f16x16;
typedef __attribute__((ext_vector_type(8)))  float    f32x8;
typedef __attribute__((ext_vector_type(4)))  float    f32x4;
typedef __attribute__((ext_vector_type(4)))  unsigned int u32x4;
typedef __attribute__((ext_vector_type(8)))  int      i32x8;
typedef __attribute__((ext_vector_type(4)))  int      i32x4;

#define N_ROWS      8192
#define DIMS        256
#define K_CODES     8192
#define STAGE       32                  // codebook rows per TDM stage (16 KB f16)
#define NSTAGES     (K_CODES / STAGE)   // 256
#define STAGE_HALFS (STAGE * DIMS)
#define MTILE       64                  // x-rows per workgroup (4 waves * 16)
#define TEMP_INV    100.0f              // 1 / ENTROPY_TEMPERATURE
#define VQ_EPS      1e-5f

union AFrag { f16x16 v; _Float16 h[16]; };
union CFrag { f32x8  v; float    f[8];  };

// Issue one TDM load: 32x256 f16 tile of the codebook -> LDS.
// D# per CDNA5 ISA ch.8: group0 {count, lds_addr, global_addr, type=2},
// group1 {data_size=2B, tensor dims/strides, tile dims}, groups 2/3 zero (2-D).
// This toolchain exposes the 6-arg builtin (extra trailing i32x8 group, unused).
__device__ __forceinline__ void tdm_load_stage(const _Float16* gsrc, unsigned lds_off) {
    unsigned long long ga = (unsigned long long)(size_t)gsrc;
    u32x4 g0;
    g0[0] = 1u;                                           // count=1, user D#
    g0[1] = lds_off;                                      // lds_addr (bytes)
    g0[2] = (unsigned)ga;                                 // global_addr[31:0]
    g0[3] = (unsigned)((ga >> 32) & 0x01ffffffu) | (2u << 30);  // addr[56:32]|type=2
    i32x8 g1;
    g1[0] = (int)(1u << 16);                              // data_size=1 -> 2 bytes
    g1[1] = (int)((unsigned)DIMS << 16);                  // tensor_dim0 = 256
    g1[2] = (int)(((unsigned)K_CODES & 0xffffu) << 16);   // tensor_dim1 = 8192
    g1[3] = (int)((unsigned)DIMS << 16);                  // tile_dim0 = 256
    g1[4] = STAGE;                                        // tile_dim1 = 32, tile_dim2=0
    g1[5] = DIMS;                                         // tensor_dim0_stride = 256
    g1[6] = 0;                                            // stride0 hi | stride1 lo16
    g1[7] = (int)(((unsigned long long)DIMS * K_CODES) >> 16);  // stride1[47:16]
    i32x4 gz4  = {0, 0, 0, 0};
    i32x8 gz8  = {0, 0, 0, 0, 0, 0, 0, 0};
    __builtin_amdgcn_tensor_load_to_lds(g0, g1, gz4, gz4, gz8, 0);
}

// Precompute f16 codebook + ||c||^2, zero accumulators. One block per code row.
__global__ __launch_bounds__(256) void vq_prep(const float* __restrict__ cb,
                                               _Float16* __restrict__ cbh,
                                               float* __restrict__ g_c2,
                                               float* __restrict__ g_avgp,
                                               float* __restrict__ g_mse) {
    __shared__ float red[8];
    const int r = blockIdx.x, t = threadIdx.x;
    const float v = cb[(size_t)r * DIMS + t];
    cbh[(size_t)r * DIMS + t] = (_Float16)v;
    float sq = v * v;
#pragma unroll
    for (int mask = 1; mask < 32; mask <<= 1) sq += __shfl_xor(sq, mask, 32);
    if ((t & 31) == 0) red[t >> 5] = sq;
    __syncthreads();
    if (t == 0) {
        float c = 0.f;
        for (int i = 0; i < 8; ++i) c += red[i];
        g_c2[r] = c;
        g_avgp[r] = 0.f;
        if (r == 0) g_mse[0] = 0.f;
    }
}

__global__ __launch_bounds__(128) void vq_main(const float* __restrict__ x,
                                               const _Float16* __restrict__ cbh,
                                               const float* __restrict__ g_c2,
                                               float* __restrict__ g_avgp,
                                               float* __restrict__ g_sent,
                                               int*   __restrict__ g_idx,
                                               float* __restrict__ out_idx_f) {
    __shared__ __align__(32) _Float16 s_cb[2 * STAGE_HALFS];   // 2 x 16 KB
    __shared__ float s_avgp[STAGE];

    const int tid  = threadIdx.x;
    const int wave = tid >> 5;
    const int lane = tid & 31;
    const int l15  = lane & 15;
    const int half = lane >> 4;
    const int rowBase = blockIdx.x * MTILE + wave * 16;
    const int myRow   = rowBase + l15;

    // -------- resident A fragments (x rows, f32 -> f16) + ||x||^2 ------------
    AFrag a[8];
    float a2 = 0.f;
    const float* xr = x + (size_t)myRow * DIMS;
#pragma unroll
    for (int j = 0; j < 8; ++j) {
        const int k0 = j * 32 + half * 16;         // contiguous k order (k = 32j + 16*half + s), same as B
        f32x4 f0 = *(const f32x4*)(xr + k0);
        f32x4 f1 = *(const f32x4*)(xr + k0 + 4);
        f32x4 f2 = *(const f32x4*)(xr + k0 + 8);
        f32x4 f3 = *(const f32x4*)(xr + k0 + 12);
#pragma unroll
        for (int i = 0; i < 4; ++i) {
            a[j].h[i]      = (_Float16)f0[i];
            a[j].h[4 + i]  = (_Float16)f1[i];
            a[j].h[8 + i]  = (_Float16)f2[i];
            a[j].h[12 + i] = (_Float16)f3[i];
            a2 += f0[i]*f0[i] + f1[i]*f1[i] + f2[i]*f2[i] + f3[i]*f3[i];
        }
    }
    a2 += __shfl_xor(a2, 16, 32);
    float a2e[8];
#pragma unroll
    for (int e = 0; e < 8; ++e) a2e[e] = __shfl(a2, half * 8 + e, 32);

    float m[8], s[8], t[8], lse[8]; int ib[8];
#pragma unroll
    for (int e = 0; e < 8; ++e) { m[e] = -1e30f; s[e] = 0.f; t[e] = 0.f; lse[e] = 0.f; ib[e] = 0; }

    if (tid < STAGE) s_avgp[tid] = 0.f;
    const unsigned lds_base = (unsigned)(unsigned long long)(size_t)(&s_cb[0]);

    // pass 0: online softmax (m,s,t) + argmin.  pass 1: p=exp(z-lse) -> avg_probs.
    for (int pass = 0; pass < 2; ++pass) {
        if (wave == 0) tdm_load_stage(cbh, lds_base);              // prime stage 0
        for (int it = 0; it < NSTAGES; ++it) {
            if (wave == 0) {
                if (it + 1 < NSTAGES) {
                    tdm_load_stage(cbh + (size_t)(it + 1) * STAGE_HALFS,
                                   lds_base + (unsigned)(((it + 1) & 1) * (STAGE_HALFS * 2)));
                    __builtin_amdgcn_s_wait_tensorcnt(1);          // stage `it` landed
                } else {
                    __builtin_amdgcn_s_wait_tensorcnt(0);
                }
            }
            __syncthreads();                                       // publish buffer it&1
            const _Float16* buf = s_cb + (it & 1) * STAGE_HALFS;
#pragma unroll
            for (int nt = 0; nt < STAGE / 16; ++nt) {
                const int nl = nt * 16 + l15;
                CFrag acc;
#pragma unroll
                for (int e = 0; e < 8; ++e) acc.f[e] = 0.f;
                const _Float16* bp = buf + nl * DIMS + half * 16;
#pragma unroll
                for (int j = 0; j < 8; ++j) {
                    f16x16 b = *(const f16x16*)(bp + j * 32);
                    acc.v = __builtin_amdgcn_wmma_f32_16x16x32_f16(
                                false, a[j].v, false, b, (short)0, acc.v, false, false);
                }
                const int   col = it * STAGE + nl;
                const float c2v = g_c2[col];
                if (pass == 0) {
#pragma unroll
                    for (int e = 0; e < 8; ++e) {
                        float z = (2.f * acc.f[e] - a2e[e] - c2v) * TEMP_INV;  // -dist/T
                        if (z > m[e]) ib[e] = col;                             // argmin dist
                        float M  = fmaxf(m[e], z);
                        float em = __expf(m[e] - M), ez = __expf(z - M);
                        s[e] = s[e] * em + ez;
                        t[e] = t[e] * em + ez * z;
                        m[e] = M;
                    }
                } else {
                    float psum = 0.f;
#pragma unroll
                    for (int e = 0; e < 8; ++e) {
                        float z = (2.f * acc.f[e] - a2e[e] - c2v) * TEMP_INV;
                        psum += __expf(z - lse[e]);
                    }
                    atomicAdd(&s_avgp[nl], psum);                  // ds_add_f32
                }
            }
            __syncthreads();                                       // buffer reuse safety
            if (pass == 1 && tid < STAGE) {
                atomicAdd(&g_avgp[it * STAGE + tid], s_avgp[tid]);
                s_avgp[tid] = 0.f;
            }
        }

        if (pass == 0) {
            // merge partials across the 16 lanes of each half-wave
#pragma unroll
            for (int e = 0; e < 8; ++e) {
#pragma unroll
                for (int mask = 1; mask <= 8; mask <<= 1) {
                    float m2 = __shfl_xor(m[e], mask, 32);
                    float s2 = __shfl_xor(s[e], mask, 32);
                    float t2 = __shfl_xor(t[e], mask, 32);
                    int   i2 = __shfl_xor(ib[e], mask, 32);
                    float M  = fmaxf(m[e], m2);
                    float e1 = __expf(m[e] - M), e2 = __expf(m2 - M);
                    s[e] = s[e] * e1 + s2 * e2;
                    t[e] = t[e] * e1 + t2 * e2;
                    if (m2 > m[e]) ib[e] = i2;
                    m[e] = M;
                }
            }
#pragma unroll
            for (int e = 0; e < 8; ++e) lse[e] = m[e] + __logf(s[e]);
            if (l15 == 0) {                              // lanes 0 and 16 own rows
#pragma unroll
                for (int e = 0; e < 8; ++e) {
                    const int row = rowBase + half * 8 + e;
                    g_idx[row]     = ib[e];
                    out_idx_f[row] = (float)ib[e];
                    g_sent[row]    = lse[e] - t[e] / s[e];   // row sample entropy
                }
            }
        }
    }
}

// quantized = codebook[idx] (forward value of straight-through output),
// fused (q - x)^2 reduction.
__global__ __launch_bounds__(256) void vq_gather(const float* __restrict__ x,
                                                 const float* __restrict__ cb,
                                                 const int* __restrict__ g_idx,
                                                 float* __restrict__ outq,
                                                 float* __restrict__ g_mse) {
    const int row = blockIdx.x;
    const int t   = threadIdx.x;
    const int code = g_idx[row];
    const float q  = cb[(size_t)code * DIMS + t];
    const float xv = x[(size_t)row * DIMS + t];
    outq[(size_t)row * DIMS + t] = q;
    float d = (q - xv) * (q - xv);
#pragma unroll
    for (int mask = 1; mask < 32; mask <<= 1) d += __shfl_xor(d, mask, 32);
    if ((t & 31) == 0) atomicAdd(g_mse, d);
}

__global__ __launch_bounds__(256) void vq_final(const float* __restrict__ g_avgp,
                                                const float* __restrict__ g_sent,
                                                const float* __restrict__ g_mse,
                                                float* __restrict__ loss_out) {
    __shared__ float red[16];
    const int t = threadIdx.x;
    const float invN = 1.0f / (float)N_ROWS;
    float aent = 0.f, sent = 0.f;
    for (int k = t; k < K_CODES; k += 256) {        // N_ROWS == K_CODES
        float avg = g_avgp[k] * invN;
        aent -= avg * __logf(avg + VQ_EPS);
        sent += g_sent[k];
    }
#pragma unroll
    for (int mask = 1; mask < 32; mask <<= 1) {
        aent += __shfl_xor(aent, mask, 32);
        sent += __shfl_xor(sent, mask, 32);
    }
    const int w = t >> 5;
    if ((t & 31) == 0) { red[w] = aent; red[8 + w] = sent; }
    __syncthreads();
    if (t == 0) {
        float A = 0.f, S = 0.f;
        for (int i = 0; i < 8; ++i) { A += red[i]; S += red[8 + i]; }
        S *= invN;
        const float mse = g_mse[0] / (float)((size_t)N_ROWS * DIMS);
        const float entropy_loss = (S - A) * 0.1f;  // ENTROPY_LOSS_RATIO
        loss_out[0] = 1.25f * mse + entropy_loss;   // (0.25 + 1.0) * mse + ent
    }
}

extern "C" void kernel_launch(void* const* d_in, const int* in_sizes, int n_in,
                              void* d_out, int out_size, void* d_ws, size_t ws_size,
                              hipStream_t stream) {
    const float* x  = (const float*)d_in[0];
    const float* cb = (const float*)d_in[1];

    float* out     = (float*)d_out;
    float* outq    = out;                              // 8192*256 quantized_st
    float* loss    = out + (size_t)N_ROWS * DIMS;      // 1 scalar loss
    float* out_idx = loss + 1;                         // 8192 indices (as f32)

    _Float16* g_cbh = (_Float16*)d_ws;                 // 8192*256 f16 codebook (4 MB)
    float* g_c2   = (float*)(g_cbh + (size_t)K_CODES * DIMS);  // 8192 ||c||^2
    float* g_avgp = g_c2 + K_CODES;                    // 8192 sum_n p[n,k]
    float* g_sent = g_avgp + K_CODES;                  // 8192 row sample entropy
    int*   g_idx  = (int*)(g_sent + N_ROWS);           // 8192 argmin indices
    float* g_mse  = (float*)(g_idx + N_ROWS);          // 1 sum (q-x)^2

    vq_prep  <<<K_CODES, 256, 0, stream>>>(cb, g_cbh, g_c2, g_avgp, g_mse);
    vq_main  <<<N_ROWS / MTILE, 128, 0, stream>>>(x, g_cbh, g_c2, g_avgp, g_sent,
                                                  g_idx, out_idx);
    vq_gather<<<N_ROWS, 256, 0, stream>>>(x, cb, g_idx, outq, g_mse);
    vq_final <<<1, 256, 0, stream>>>(g_avgp, g_sent, g_mse, loss);
}
